// NoiseConditionalProteinMPNN_83829171684070
// MI455X (gfx1250) — compile-verified, matrix operand order checked by reading, not run on hardware
//
#include <hip/hip_runtime.h>
#include <math.h>

// ---------------- constants ----------------
constexpr int HD   = 128;   // hidden
constexpr int KNN  = 32;    // neighbors
constexpr int Bb   = 16;    // batch
constexpr int LL   = 512;   // seq len
constexpr int NBN  = Bb * LL;         // nodes
constexpr int TCD  = 512;   // time cond dim
constexpr float SCALE_INV = 1.0f / 30.0f;
constexpr float EPS_LN = 1e-5f;

typedef __bf16 bf16;
typedef __attribute__((ext_vector_type(16))) __bf16 bf16x16;
typedef __attribute__((ext_vector_type(8)))  float  floatx8;

__device__ inline bf16 f2bf(float x) { return (bf16)x; }
__device__ inline float gelu_exact(float x) {
  return 0.5f * x * (1.0f + erff(x * 0.70710678118654752440f));
}
__device__ inline float silu_f(float x) { return x / (1.0f + expf(-x)); }

// atom ids: 0=N 1=Ca 2=C 3=O 4=Cb
__constant__ int PAIR_A[25] = {1,0,2,3,4, 1,1,1,1, 0,0,0, 4,4, 3, 0,2,3,4, 2,3,4, 2,3, 2};
__constant__ int PAIR_B[25] = {1,0,2,3,4, 0,2,3,4, 2,3,4, 2,3, 2, 1,1,1,1, 0,0,0, 4,4, 3};

// ---------------- WMMA helpers ----------------
union FragU { bf16x16 v; uint4 q[2]; };

__device__ inline bf16x16 load_frag(const bf16* lo, const bf16* hi) {
  FragU f;
  f.q[0] = *(const uint4*)lo;
  f.q[1] = *(const uint4*)hi;
  return f.v;
}

// C = A(32 x Kd tile rows mbase..mbase+15) * Wt^T  for one 16x16 N-tile.
// A: row-major bf16, leading dim lda. Wt: [N][Kd] bf16 (pre-transposed weights).
__device__ inline floatx8 gemm32(const bf16* __restrict__ A, int lda,
                                 const bf16* __restrict__ Wt, int Kd,
                                 int mbase, int nbase, int lane,
                                 const float* __restrict__ bias) {
  const int half = lane >> 4;
  const bf16* arow = A + (size_t)(mbase + (lane & 15)) * lda + half * 8;
  const bf16* brow = Wt + (size_t)(nbase + (lane & 15)) * Kd + half * 8;
  floatx8 acc;
  float bv = bias ? bias[nbase + (lane & 15)] : 0.0f;
#pragma unroll
  for (int v = 0; v < 8; ++v) acc[v] = bv;
  for (int k = 0; k < Kd; k += 32) {
    bf16x16 af = load_frag(arow, arow + 16);
    bf16x16 bfv = load_frag(brow, brow + 16);
    acc = __builtin_amdgcn_wmma_f32_16x16x32_bf16(false, af, false, bfv,
                                                  (short)0, acc, false, false);
    arow += 32; brow += 32;
  }
  return acc;
}

// store tile with exact-GELU -> bf16 buffer (row-major, leading dim ldo)
__device__ inline void epi_gelu_bf16(bf16* Out, int ldo, int mbase, int nbase,
                                     int lane, floatx8 acc) {
  const int half = lane >> 4;
  const int n = nbase + (lane & 15);
#pragma unroll
  for (int v = 0; v < 8; ++v)
    Out[(size_t)(mbase + 8 * half + v) * ldo + n] = f2bf(gelu_exact(acc[v]));
}

// ---------------- small kernels ----------------
__global__ void k_transpose_bf16(const float* __restrict__ W, bf16* __restrict__ Wt,
                                 int Kd, int Nd) {
  int idx = blockIdx.x * blockDim.x + threadIdx.x;
  if (idx >= Kd * Nd) return;
  int n = idx / Kd, k = idx - n * Kd;
  Wt[idx] = f2bf(W[(size_t)k * Nd + n]);
}

__global__ void k_zero(float* p, int n) {
  int i = blockIdx.x * blockDim.x + threadIdx.x;
  if (i < n) p[i] = 0.0f;
}

__global__ void k_copy(const float* __restrict__ s, float* __restrict__ d, int n) {
  int i = blockIdx.x * blockDim.x + threadIdx.x;
  if (i < n) d[i] = s[i];
}

__global__ void k_hs(const int* __restrict__ aatype, const float* __restrict__ Ws,
                     bf16* __restrict__ hS) {
  int idx = blockIdx.x * blockDim.x + threadIdx.x;
  if (idx >= NBN * HD) return;
  int h = idx & (HD - 1);
  int n = idx >> 7;
  hS[idx] = f2bf(Ws[(size_t)aatype[n] * HD + h]);
}

__global__ void k_cb(const float* __restrict__ coords, float* __restrict__ Cb) {
  int n = blockIdx.x * blockDim.x + threadIdx.x;
  if (n >= NBN) return;
  const float* c = coords + (size_t)n * 12;
  float Nx = c[0], Ny = c[1], Nz = c[2];
  float Ax = c[3], Ay = c[4], Az = c[5];
  float Cx = c[6], Cy = c[7], Cz = c[8];
  float bx = Ax - Nx, by = Ay - Ny, bz = Az - Nz;
  float cx = Cx - Ax, cy = Cy - Ay, cz = Cz - Az;
  float axv = by * cz - bz * cy;
  float ayv = bz * cx - bx * cz;
  float azv = bx * cy - by * cx;
  Cb[n * 3 + 0] = -0.58273431f * axv + 0.56802827f * bx - 0.54067466f * cx + Ax;
  Cb[n * 3 + 1] = -0.58273431f * ayv + 0.56802827f * by - 0.54067466f * cy + Ay;
  Cb[n * 3 + 2] = -0.58273431f * azv + 0.56802827f * bz - 0.54067466f * cz + Az;
}

// FiLM conditioning: silu(tc) @ time.w + time.b -> [B][256] (sc | sh)
__global__ void k_film(const float* __restrict__ tc, const float* __restrict__ tw,
                       const float* __restrict__ tb, float* __restrict__ film) {
  int b = blockIdx.x, n = threadIdx.x;  // n < 256
  float acc = tb[n];
  const float* t = tc + (size_t)b * TCD;
  for (int k = 0; k < TCD; ++k) acc += silu_f(t[k]) * tw[(size_t)k * 256 + n];
  film[b * 256 + n] = acc;
}

// ---------------- KNN ----------------
__global__ void __launch_bounds__(256)
k_knn(const float* __restrict__ coords, const float* __restrict__ mask,
      int* __restrict__ eidx) {
  __shared__ float sd[LL];
  __shared__ float redv[256];
  __shared__ int   redi[256];
  int node = blockIdx.x;
  int b = node / LL, i = node - b * LL;
  int tid = threadIdx.x;
  const float* cb = coords + (size_t)(b * LL) * 12;
  float mi = mask[node];
  float xi = cb[i * 12 + 3], yi = cb[i * 12 + 4], zi = cb[i * 12 + 5];
  for (int j = tid; j < LL; j += 256) {
    float dx = xi - cb[j * 12 + 3];
    float dy = yi - cb[j * 12 + 4];
    float dz = zi - cb[j * 12 + 5];
    float m2 = mi * mask[b * LL + j];
    sd[j] = m2 * sqrtf(dx * dx + dy * dy + dz * dz + 1e-6f);
  }
  __syncthreads();
  float mx = -1e30f;
  for (int j = tid; j < LL; j += 256) mx = fmaxf(mx, sd[j]);
  redv[tid] = mx; __syncthreads();
  for (int s = 128; s > 0; s >>= 1) {
    if (tid < s) redv[tid] = fmaxf(redv[tid], redv[tid + s]);
    __syncthreads();
  }
  float rowmax = redv[0]; __syncthreads();
  for (int j = tid; j < LL; j += 256) {
    float m2 = mi * mask[b * LL + j];
    sd[j] = sd[j] + (1.0f - m2) * rowmax;
  }
  __syncthreads();
  for (int k = 0; k < KNN; ++k) {
    float bv = 1e30f; int bi = LL;
    for (int j = tid; j < LL; j += 256) {
      float v = sd[j];
      if (v < bv || (v == bv && j < bi)) { bv = v; bi = j; }
    }
    redv[tid] = bv; redi[tid] = bi; __syncthreads();
    for (int s = 128; s > 0; s >>= 1) {
      if (tid < s) {
        float ov = redv[tid + s]; int oi = redi[tid + s];
        if (ov < redv[tid] || (ov == redv[tid] && oi < redi[tid])) {
          redv[tid] = ov; redi[tid] = oi;
        }
      }
      __syncthreads();
    }
    if (tid == 0) { eidx[node * KNN + k] = redi[0]; sd[redi[0]] = 1e30f; }
    __syncthreads();
  }
}

// ---------------- edge embedding (features -> 416 GEMM -> LN -> W_e) ----------
__device__ inline void get_atom(const float* coords, const float* Cb, int b,
                                int l, int atom, float& x, float& y, float& z) {
  if (atom == 4) {
    const float* p = Cb + (size_t)(b * LL + l) * 3;
    x = p[0]; y = p[1]; z = p[2];
  } else {
    const float* p = coords + ((size_t)(b * LL + l) * 4 + atom) * 3;
    x = p[0]; y = p[1]; z = p[2];
  }
}

__global__ void __launch_bounds__(128)
k_edge_embed(const float* __restrict__ coords, const float* __restrict__ Cb,
             const int* __restrict__ eidx, const float* __restrict__ mask,
             const int* __restrict__ ridx,
             const float* __restrict__ posw, const float* __restrict__ posb,
             const bf16* __restrict__ wt_edge,
             const float* __restrict__ ne_g, const float* __restrict__ ne_b,
             const bf16* __restrict__ wt_we, const float* __restrict__ we_b,
             float* __restrict__ hE) {
  __shared__ __align__(16) bf16  feat[32 * 416];
  __shared__ __align__(16) float etile[32 * 128];
  __shared__ __align__(16) bf16  abuf[32 * 128];
  __shared__ float mrow[32], vrow[32];

  int node = blockIdx.x;
  int b = node / LL, i = node - b * LL;
  int tid = threadIdx.x;

  // RBF features (25 pairs x 16 gaussians)
  for (int w = tid; w < 32 * 25; w += 128) {
    int k = w / 25, p = w - k * 25;
    int j = eidx[node * KNN + k];
    float ax, ay, az, bx, by, bz;
    get_atom(coords, Cb, b, i, PAIR_A[p], ax, ay, az);
    get_atom(coords, Cb, b, j, PAIR_B[p], bx, by, bz);
    float dx = ax - bx, dy = ay - by, dz = az - bz;
    float d = sqrtf(dx * dx + dy * dy + dz * dz + 1e-6f);
#pragma unroll
    for (int t = 0; t < 16; ++t) {
      float mu = 2.0f + (20.0f / 15.0f) * t;
      float zz = (d - mu) * (1.0f / 1.25f);
      feat[k * 416 + 16 + p * 16 + t] = f2bf(__expf(-zz * zz));
    }
  }
  // positional features (one-hot @ pos.w + pos.b == row select)
  for (int w = tid; w < 32 * 16; w += 128) {
    int k = w >> 4, t = w & 15;
    int j = eidx[node * KNN + k];
    int off = ridx[node] - ridx[b * LL + j];
    int dch = (mask[node] == mask[b * LL + j]) ? 1 : 0;
    int pe = dch ? min(max(off + 32, 0), 64) : 65;
    feat[k * 416 + t] = f2bf(posw[pe * 16 + t] + posb[t]);
  }
  __syncthreads();

  int wv = tid >> 5, lane = tid & 31;
  int mbase = (wv & 1) * 16;
  int nb0 = (wv >> 1) * 64;
  // GEMM 416 -> 128 (no bias) -> f32 tile
  for (int t = 0; t < 4; ++t) {
    int nbase = nb0 + t * 16;
    floatx8 acc = gemm32(feat, 416, wt_edge, 416, mbase, nbase, lane, nullptr);
    int half = lane >> 4, n = nbase + (lane & 15);
#pragma unroll
    for (int v = 0; v < 8; ++v) etile[(mbase + 8 * half + v) * HD + n] = acc[v];
  }
  __syncthreads();
  // per-edge LayerNorm (norm_edges)
  if (tid < 32) {
    float s = 0.0f, q = 0.0f;
    for (int c = 0; c < HD; ++c) { float x = etile[tid * HD + c]; s += x; q += x * x; }
    float m = s * (1.0f / HD);
    mrow[tid] = m;
    vrow[tid] = q * (1.0f / HD) - m * m;
  }
  __syncthreads();
  for (int w = tid; w < 32 * HD; w += 128) {
    int r = w >> 7, c = w & (HD - 1);
    float rs = rsqrtf(vrow[r] + EPS_LN);
    abuf[w] = f2bf((etile[w] - mrow[r]) * rs * ne_g[c] + ne_b[c]);
  }
  __syncthreads();
  // W_e: 128 -> 128 + bias -> h_E
  for (int t = 0; t < 4; ++t) {
    int nbase = nb0 + t * 16;
    floatx8 acc = gemm32(abuf, HD, wt_we, HD, mbase, nbase, lane, we_b);
    int half = lane >> 4, n = nbase + (lane & 15);
#pragma unroll
    for (int v = 0; v < 8; ++v) {
      int m = mbase + 8 * half + v;
      hE[((size_t)node * KNN + m) * HD + n] = acc[v];
    }
  }
}

// ---------------- encoder node message ----------------
__global__ void __launch_bounds__(128)
k_enc_node_msg(const float* __restrict__ hV, const float* __restrict__ hE,
               const int* __restrict__ eidx, const float* __restrict__ mask,
               const float* __restrict__ film,
               const bf16* __restrict__ wt1, const float* __restrict__ b1,
               const bf16* __restrict__ wt2, const float* __restrict__ b2,
               const bf16* __restrict__ wt3, const float* __restrict__ b3,
               const float* __restrict__ n1g, const float* __restrict__ n1b,
               float* __restrict__ hvmid) {
  __shared__ __align__(16) bf16  A0[32 * 384];
  __shared__ __align__(16) bf16  A1[32 * 128];
  __shared__ __align__(16) float Mt[32 * 128];
  __shared__ float hvi[128];
  __shared__ float matt[32];
  __shared__ float redA[128], redB[128];

  int node = blockIdx.x;
  int b = node / LL;
  int tid = threadIdx.x;
  const float* sc = film + b * 256;
  const float* sh = sc + 128;
  float mi = mask[node];

  if (tid < 128) hvi[tid] = hV[(size_t)node * HD + tid] * (1.0f + sc[tid]) + sh[tid];
  if (tid < 32) {
    int j = eidx[node * KNN + tid];
    matt[tid] = mi * mask[b * LL + j];
  }
  __syncthreads();

  for (int w = tid; w < 32 * 384; w += 128) {
    int r = w / 384, c = w - r * 384;
    float v;
    if (c < 128) v = hvi[c];
    else if (c < 256) v = hE[((size_t)node * KNN + r) * HD + (c - 128)];
    else {
      int j = eidx[node * KNN + r];
      int jn = b * LL + j;
      int cc = c - 256;
      v = hV[(size_t)jn * HD + cc] * (1.0f + sc[cc]) + sh[cc];
    }
    A0[w] = f2bf(v);
  }
  __syncthreads();

  int wv = tid >> 5, lane = tid & 31;
  int mbase = (wv & 1) * 16;
  int nb0 = (wv >> 1) * 64;

  for (int t = 0; t < 4; ++t) {   // W1: 384 -> 128, gelu
    int nbase = nb0 + t * 16;
    floatx8 acc = gemm32(A0, 384, wt1, 384, mbase, nbase, lane, b1);
    epi_gelu_bf16(A1, 128, mbase, nbase, lane, acc);
  }
  __syncthreads();
  for (int t = 0; t < 4; ++t) {   // W2: 128 -> 128, gelu (into A0 region, lda 384)
    int nbase = nb0 + t * 16;
    floatx8 acc = gemm32(A1, 128, wt2, 128, mbase, nbase, lane, b2);
    epi_gelu_bf16(A0, 384, mbase, nbase, lane, acc);
  }
  __syncthreads();
  for (int t = 0; t < 4; ++t) {   // W3: 128 -> 128, * mask_attend
    int nbase = nb0 + t * 16;
    floatx8 acc = gemm32(A0, 384, wt3, 128, mbase, nbase, lane, b3);
    int half = lane >> 4, n = nbase + (lane & 15);
#pragma unroll
    for (int v = 0; v < 8; ++v) {
      int m = mbase + 8 * half + v;
      Mt[m * HD + n] = acc[v] * matt[m];
    }
  }
  __syncthreads();

  float h = 0.0f;
  if (tid < 128) {
    float s = 0.0f;
    for (int r = 0; r < 32; ++r) s += Mt[r * HD + tid];
    h = hvi[tid] + s * SCALE_INV;
    redA[tid] = h; redB[tid] = h * h;
  }
  __syncthreads();
  for (int s2 = 64; s2 > 0; s2 >>= 1) {
    if (tid < s2) { redA[tid] += redA[tid + s2]; redB[tid] += redB[tid + s2]; }
    __syncthreads();
  }
  float mean = redA[0] * (1.0f / HD);
  float var = redB[0] * (1.0f / HD) - mean * mean;
  float rs = rsqrtf(var + EPS_LN);
  if (tid < 128)
    hvmid[(size_t)node * HD + tid] = (h - mean) * rs * n1g[tid] + n1b[tid];
}

// ---------------- encoder edge update ----------------
__global__ void __launch_bounds__(128)
k_enc_edge_msg(const float* __restrict__ hV, float* __restrict__ hE,
               const int* __restrict__ eidx,
               const bf16* __restrict__ wt11, const float* __restrict__ b11,
               const bf16* __restrict__ wt12, const float* __restrict__ b12,
               const bf16* __restrict__ wt13, const float* __restrict__ b13,
               const float* __restrict__ n3g, const float* __restrict__ n3b) {
  __shared__ __align__(16) bf16  A0[32 * 384];
  __shared__ __align__(16) bf16  A1[32 * 128];
  __shared__ __align__(16) float Mt[32 * 128];

  int node = blockIdx.x;
  int b = node / LL;
  int tid = threadIdx.x;

  for (int w = tid; w < 32 * 384; w += 128) {
    int r = w / 384, c = w - r * 384;
    float v;
    if (c < 128) v = hV[(size_t)node * HD + c];
    else if (c < 256) v = hE[((size_t)node * KNN + r) * HD + (c - 128)];
    else {
      int j = eidx[node * KNN + r];
      v = hV[(size_t)(b * LL + j) * HD + (c - 256)];
    }
    A0[w] = f2bf(v);
  }
  __syncthreads();

  int wv = tid >> 5, lane = tid & 31;
  int mbase = (wv & 1) * 16;
  int nb0 = (wv >> 1) * 64;

  for (int t = 0; t < 4; ++t) {
    int nbase = nb0 + t * 16;
    floatx8 acc = gemm32(A0, 384, wt11, 384, mbase, nbase, lane, b11);
    epi_gelu_bf16(A1, 128, mbase, nbase, lane, acc);
  }
  __syncthreads();
  for (int t = 0; t < 4; ++t) {
    int nbase = nb0 + t * 16;
    floatx8 acc = gemm32(A1, 128, wt12, 128, mbase, nbase, lane, b12);
    epi_gelu_bf16(A0, 384, mbase, nbase, lane, acc);
  }
  __syncthreads();
  for (int t = 0; t < 4; ++t) {
    int nbase = nb0 + t * 16;
    floatx8 acc = gemm32(A0, 384, wt13, 128, mbase, nbase, lane, b13);
    int half = lane >> 4, n = nbase + (lane & 15);
#pragma unroll
    for (int v = 0; v < 8; ++v) {
      int m = mbase + 8 * half + v;
      Mt[m * HD + n] = acc[v] + hE[((size_t)node * KNN + m) * HD + n];
    }
  }
  __syncthreads();
  if (tid < 32) {  // per-edge LN(n3) and write back
    float s = 0.0f, q = 0.0f;
    for (int c = 0; c < HD; ++c) { float x = Mt[tid * HD + c]; s += x; q += x * x; }
    float m = s * (1.0f / HD);
    float var = q * (1.0f / HD) - m * m;
    float rs = rsqrtf(var + EPS_LN);
    for (int c = 0; c < HD; ++c)
      hE[((size_t)node * KNN + tid) * HD + c] = (Mt[tid * HD + c] - m) * rs * n3g[c] + n3b[c];
  }
}

// ---------------- decoder node message ----------------
__global__ void __launch_bounds__(128)
k_dec_node_msg(const float* __restrict__ hV, const float* __restrict__ hE,
               const bf16* __restrict__ hS, const int* __restrict__ eidx,
               const float* __restrict__ film,
               const bf16* __restrict__ wt1, const float* __restrict__ b1,
               const bf16* __restrict__ wt2, const float* __restrict__ b2,
               const bf16* __restrict__ wt3, const float* __restrict__ b3,
               const float* __restrict__ n1g, const float* __restrict__ n1b,
               float* __restrict__ hvmid) {
  __shared__ __align__(16) bf16  A0[32 * 512];
  __shared__ __align__(16) bf16  A1[32 * 128];
  __shared__ __align__(16) float Mt[32 * 128];
  __shared__ float hvi[128];
  __shared__ float redA[128], redB[128];

  int node = blockIdx.x;
  int b = node / LL;
  int tid = threadIdx.x;
  const float* sc = film + b * 256;
  const float* sh = sc + 128;

  if (tid < 128) hvi[tid] = hV[(size_t)node * HD + tid] * (1.0f + sc[tid]) + sh[tid];
  __syncthreads();

  for (int w = tid; w < 32 * 512; w += 128) {
    int r = w >> 9, c = w & 511;
    int j = eidx[node * KNN + r];
    int jn = b * LL + j;
    bf16 v;
    if (c < 128) v = f2bf(hvi[c]);
    else if (c < 256) v = f2bf(hE[((size_t)node * KNN + r) * HD + (c - 128)]);
    else if (c < 384) v = hS[(size_t)jn * HD + (c - 256)];
    else v = f2bf(hV[(size_t)jn * HD + (c - 384)]);
    A0[w] = v;
  }
  __syncthreads();

  int wv = tid >> 5, lane = tid & 31;
  int mbase = (wv & 1) * 16;
  int nb0 = (wv >> 1) * 64;

  for (int t = 0; t < 4; ++t) {   // W1: 512 -> 128
    int nbase = nb0 + t * 16;
    floatx8 acc = gemm32(A0, 512, wt1, 512, mbase, nbase, lane, b1);
    epi_gelu_bf16(A1, 128, mbase, nbase, lane, acc);
  }
  __syncthreads();
  for (int t = 0; t < 4; ++t) {   // W2 (into A0 region, lda 512)
    int nbase = nb0 + t * 16;
    floatx8 acc = gemm32(A1, 128, wt2, 128, mbase, nbase, lane, b2);
    epi_gelu_bf16(A0, 512, mbase, nbase, lane, acc);
  }
  __syncthreads();
  for (int t = 0; t < 4; ++t) {   // W3
    int nbase = nb0 + t * 16;
    floatx8 acc = gemm32(A0, 512, wt3, 128, mbase, nbase, lane, b3);
    int half = lane >> 4, n = nbase + (lane & 15);
#pragma unroll
    for (int v = 0; v < 8; ++v) {
      int m = mbase + 8 * half + v;
      Mt[m * HD + n] = acc[v];
    }
  }
  __syncthreads();

  float h = 0.0f;
  if (tid < 128) {
    float s = 0.0f;
    for (int r = 0; r < 32; ++r) s += Mt[r * HD + tid];
    h = hvi[tid] + s * SCALE_INV;
    redA[tid] = h; redB[tid] = h * h;
  }
  __syncthreads();
  for (int s2 = 64; s2 > 0; s2 >>= 1) {
    if (tid < s2) { redA[tid] += redA[tid + s2]; redB[tid] += redB[tid + s2]; }
    __syncthreads();
  }
  float mean = redA[0] * (1.0f / HD);
  float var = redB[0] * (1.0f / HD) - mean * mean;
  float rs = rsqrtf(var + EPS_LN);
  if (tid < 128)
    hvmid[(size_t)node * HD + tid] = (h - mean) * rs * n1g[tid] + n1b[tid];
}

// ---------------- FFN + LN(n2) + mask ----------------
__global__ void __launch_bounds__(128)
k_ff_ln(const float* __restrict__ hvmid, const float* __restrict__ mask,
        const bf16* __restrict__ wtin, const float* __restrict__ bin,
        const bf16* __restrict__ wtout, const float* __restrict__ bout,
        const float* __restrict__ n2g, const float* __restrict__ n2b,
        float* __restrict__ hVout) {
  __shared__ __align__(16) bf16  A[32 * 128];
  __shared__ __align__(16) bf16  Mid[32 * 512];
  __shared__ __align__(16) float Outs[32 * 128];
  int node0 = blockIdx.x * 32;
  int tid = threadIdx.x;

  for (int w = tid; w < 32 * 128; w += 128)
    A[w] = f2bf(hvmid[(size_t)node0 * HD + w]);
  __syncthreads();

  int wv = tid >> 5, lane = tid & 31;
  int mbase = (wv & 1) * 16;
  int nb0 = (wv >> 1) * 256;
  for (int t = 0; t < 16; ++t) {  // ff_in: 128 -> 512, gelu
    int nbase = nb0 + t * 16;
    floatx8 acc = gemm32(A, 128, wtin, 128, mbase, nbase, lane, bin);
    epi_gelu_bf16(Mid, 512, mbase, nbase, lane, acc);
  }
  __syncthreads();
  int nb1 = (wv >> 1) * 64;
  for (int t = 0; t < 4; ++t) {   // ff_out: 512 -> 128 + residual
    int nbase = nb1 + t * 16;
    floatx8 acc = gemm32(Mid, 512, wtout, 512, mbase, nbase, lane, bout);
    int half = lane >> 4, n = nbase + (lane & 15);
#pragma unroll
    for (int v = 0; v < 8; ++v) {
      int m = mbase + 8 * half + v;
      Outs[m * HD + n] = acc[v] + hvmid[(size_t)(node0 + m) * HD + n];
    }
  }
  __syncthreads();
  if (tid < 32) {  // per-node LN(n2) then * mask
    float s = 0.0f, q = 0.0f;
    for (int c = 0; c < HD; ++c) { float x = Outs[tid * HD + c]; s += x; q += x * x; }
    float m = s * (1.0f / HD);
    float var = q * (1.0f / HD) - m * m;
    float rs = rsqrtf(var + EPS_LN);
    float mk = mask[node0 + tid];
    for (int c = 0; c < HD; ++c)
      hVout[(size_t)(node0 + tid) * HD + c] =
          mk * ((Outs[tid * HD + c] - m) * rs * n2g[c] + n2b[c]);
  }
}

// ---------------- host orchestration ----------------
// d_in layout (jax pytree flatten; dicts sorted by key):
// 0 coords 1 aatype 2 mask 3 ridx 4 time_cond
// 5 W_e.b 6 W_e.w 7 W_s
// 8 + 16*l : dec layer l: W1b W1w W2b W2w W3b W3w FINb FINw FOUTb FOUTw n1b n1g n2b n2g Tb Tw
// 56 edge_emb.w
// 57 + 24*l : enc layer l: W1b W1w W11b W11w W12b W12w W13b W13w W2b W2w W3b W3w
//                          FINb FINw FOUTb FOUTw n1b n1g n2b n2g n3b n3g Tb Tw
// 129 norm_edges.b 130 norm_edges.g 131 pos.b 132 pos.w

extern "C" void kernel_launch(void* const* d_in, const int* in_sizes, int n_in,
                              void* d_out, int out_size, void* d_ws, size_t ws_size,
                              hipStream_t stream) {
  const float* coords = (const float*)d_in[0];
  const int*   aatype = (const int*)d_in[1];
  const float* maskp  = (const float*)d_in[2];
  const int*   ridx   = (const int*)d_in[3];
  const float* tc     = (const float*)d_in[4];
  const float* We_b   = (const float*)d_in[5];
  const float* We_w   = (const float*)d_in[6];
  const float* Ws     = (const float*)d_in[7];
  const float* edge_w = (const float*)d_in[56];
  const float* ne_b   = (const float*)d_in[129];
  const float* ne_g   = (const float*)d_in[130];
  const float* pos_b  = (const float*)d_in[131];
  const float* pos_w  = (const float*)d_in[132];
  auto DEC = [&](int l, int o) { return (const float*)d_in[8 + 16 * l + o]; };
  auto ENC = [&](int l, int o) { return (const float*)d_in[57 + 24 * l + o]; };

  // workspace carve-up
  char* base = (char*)d_ws;
  size_t off = 0;
  auto alloc = [&](size_t bytes) -> void* {
    void* p = base + off;
    off = (off + bytes + 255) & ~(size_t)255;
    return p;
  };
  float* Cb    = (float*)alloc((size_t)NBN * 3 * 4);
  int*   eidx  = (int*)alloc((size_t)NBN * KNN * 4);
  float* hE    = (float*)alloc((size_t)NBN * KNN * HD * 4);
  float* hV    = (float*)alloc((size_t)NBN * HD * 4);
  float* hvmid = (float*)alloc((size_t)NBN * HD * 4);
  float* film  = (float*)alloc((size_t)Bb * 256 * 4);
  bf16*  hS    = (bf16*)alloc((size_t)NBN * HD * 2);
  bf16*  wt_edge = (bf16*)alloc((size_t)128 * 416 * 2);
  bf16*  wt_we   = (bf16*)alloc((size_t)128 * 128 * 2);
  struct EncW { bf16 *w1, *w11, *w12, *w13, *w2, *w3, *fin, *fout; } encw[3];
  struct DecW { bf16 *w1, *w2, *w3, *fin, *fout; } decw[3];
  for (int l = 0; l < 3; ++l) {
    encw[l].w1   = (bf16*)alloc((size_t)128 * 384 * 2);
    encw[l].w11  = (bf16*)alloc((size_t)128 * 384 * 2);
    encw[l].w12  = (bf16*)alloc((size_t)128 * 128 * 2);
    encw[l].w13  = (bf16*)alloc((size_t)128 * 128 * 2);
    encw[l].w2   = (bf16*)alloc((size_t)128 * 128 * 2);
    encw[l].w3   = (bf16*)alloc((size_t)128 * 128 * 2);
    encw[l].fin  = (bf16*)alloc((size_t)512 * 128 * 2);
    encw[l].fout = (bf16*)alloc((size_t)128 * 512 * 2);
    decw[l].w1   = (bf16*)alloc((size_t)128 * 512 * 2);
    decw[l].w2   = (bf16*)alloc((size_t)128 * 128 * 2);
    decw[l].w3   = (bf16*)alloc((size_t)128 * 128 * 2);
    decw[l].fin  = (bf16*)alloc((size_t)512 * 128 * 2);
    decw[l].fout = (bf16*)alloc((size_t)128 * 512 * 2);
  }
  if (off > ws_size) return;  // not enough scratch

  auto T = [&](const float* src, bf16* dst, int Kd, int Nd) {
    int n = Kd * Nd;
    k_transpose_bf16<<<(n + 255) / 256, 256, 0, stream>>>(src, dst, Kd, Nd);
  };

  // ---- weight prep ----
  T(edge_w, wt_edge, 416, 128);
  T(We_w, wt_we, 128, 128);
  for (int l = 0; l < 3; ++l) {
    T(ENC(l, 1), encw[l].w1, 384, 128);
    T(ENC(l, 3), encw[l].w11, 384, 128);
    T(ENC(l, 5), encw[l].w12, 128, 128);
    T(ENC(l, 7), encw[l].w13, 128, 128);
    T(ENC(l, 9), encw[l].w2, 128, 128);
    T(ENC(l, 11), encw[l].w3, 128, 128);
    T(ENC(l, 13), encw[l].fin, 128, 512);
    T(ENC(l, 15), encw[l].fout, 512, 128);
    T(DEC(l, 1), decw[l].w1, 512, 128);
    T(DEC(l, 3), decw[l].w2, 128, 128);
    T(DEC(l, 5), decw[l].w3, 128, 128);
    T(DEC(l, 7), decw[l].fin, 128, 512);
    T(DEC(l, 9), decw[l].fout, 512, 128);
  }

  // ---- graph features ----
  k_hs<<<(NBN * HD + 255) / 256, 256, 0, stream>>>(aatype, Ws, hS);
  k_cb<<<(NBN + 255) / 256, 256, 0, stream>>>(coords, Cb);
  k_knn<<<NBN, 256, 0, stream>>>(coords, maskp, eidx);
  k_edge_embed<<<NBN, 128, 0, stream>>>(coords, Cb, eidx, maskp, ridx, pos_w,
                                        pos_b, wt_edge, ne_g, ne_b, wt_we, We_b, hE);
  k_zero<<<(NBN * HD + 255) / 256, 256, 0, stream>>>(hV, NBN * HD);

  // ---- encoder ----
  for (int l = 0; l < 3; ++l) {
    k_film<<<Bb, 256, 0, stream>>>(tc, ENC(l, 23), ENC(l, 22), film);
    k_enc_node_msg<<<NBN, 128, 0, stream>>>(
        hV, hE, eidx, maskp, film, encw[l].w1, ENC(l, 0), encw[l].w2, ENC(l, 8),
        encw[l].w3, ENC(l, 10), ENC(l, 17), ENC(l, 16), hvmid);
    k_ff_ln<<<NBN / 32, 128, 0, stream>>>(hvmid, maskp, encw[l].fin, ENC(l, 12),
                                          encw[l].fout, ENC(l, 14), ENC(l, 19),
                                          ENC(l, 18), hV);
    k_enc_edge_msg<<<NBN, 128, 0, stream>>>(hV, hE, eidx, encw[l].w11, ENC(l, 2),
                                            encw[l].w12, ENC(l, 4), encw[l].w13,
                                            ENC(l, 6), ENC(l, 21), ENC(l, 20));
  }
  // encoder embeddings -> second half of output
  k_copy<<<(NBN * HD + 255) / 256, 256, 0, stream>>>(hV, (float*)d_out + (size_t)NBN * HD,
                                                     NBN * HD);

  // ---- decoder ----
  for (int l = 0; l < 3; ++l) {
    k_film<<<Bb, 256, 0, stream>>>(tc, DEC(l, 15), DEC(l, 14), film);
    k_dec_node_msg<<<NBN, 128, 0, stream>>>(
        hV, hE, hS, eidx, film, decw[l].w1, DEC(l, 0), decw[l].w2, DEC(l, 2),
        decw[l].w3, DEC(l, 4), DEC(l, 11), DEC(l, 10), hvmid);
    k_ff_ln<<<NBN / 32, 128, 0, stream>>>(hvmid, maskp, decw[l].fin, DEC(l, 6),
                                          decw[l].fout, DEC(l, 8), DEC(l, 13),
                                          DEC(l, 12), hV);
  }
  // final h_V -> first half of output
  k_copy<<<(NBN * HD + 255) / 256, 256, 0, stream>>>(hV, (float*)d_out, NBN * HD);
}
